// LinearAttentionKernel_5299989643475
// MI455X (gfx1250) — matrix operand order repaired and checked by reading, not compiled, not run on hardware
//
#include <hip/hip_runtime.h>
#include <hip/hip_bf16.h>

// Causal linear attention (elu+1 feature map), chunked-scan formulation.
// B=4, L=4096, D=F=512.  Chunk C=32 == WMMA K for bf16 16x16x32.
// Grid: B * (D/64) = 32 workgroups, 4 wave32 each.
// Per-wave f32 state: S[512 x 16] kept in 32 WMMA accumulator tiles (256 VGPRs).

#define BATCH   4
#define SEQ     4096
#define DMODEL  512
#define CHUNK   32
#define DTILE   64
#define NWAVE   4
#define EPS_DEN 1e-8f

typedef __attribute__((ext_vector_type(16))) __bf16 v16bf;
typedef __attribute__((ext_vector_type(8)))  float  v8f;

union ABPack { unsigned int u[8]; v16bf v; };

__device__ __forceinline__ unsigned short f2bf(float x) {
  union { float f; unsigned int u; } c; c.f = x;
  unsigned int r = c.u + 0x7FFFu + ((c.u >> 16) & 1u);   // round-to-nearest-even
  return (unsigned short)(r >> 16);
}
__device__ __forceinline__ float bf2f(unsigned short h) {
  union { float f; unsigned int u; } c; c.u = ((unsigned int)h) << 16;
  return c.f;
}
__device__ __forceinline__ float phi_elu1(float x) {
  return x > 0.0f ? x + 1.0f : __expf(x);   // elu(x)+1
}

// ---- WMMA operand gathers from LDS, contiguous k order shared by A and B ----

// A-matrix 16x32 bf16: lane holds row M = row + lane%16.
// dword v: K = 2v + (lane>=16 ? 16 : 0).  storage [m][k], kstride even.
__device__ __forceinline__ v16bf load_a(const unsigned short* base, int row,
                                        int kbase, int kstride) {
  ABPack p;
  const unsigned int* b32 = (const unsigned int*)base;
  const int lane = threadIdx.x & 31;
  const int r = row + (lane & 15);
  const int hi8 = (lane & 16) ? 16 : 0;
#pragma unroll
  for (int v = 0; v < 8; ++v) {
    const int k = kbase + 2 * v + hi8;
    p.u[v] = b32[(r * kstride + k) >> 1];
  }
  return p.v;
}

// B-matrix 32x16 bf16: lane holds col N = nbase + lane%16.
// dword v: K = 2v + (lane>=16 ? 16 : 0).  storage [n][k] (i.e. B transposed), kstride even.
__device__ __forceinline__ v16bf load_b(const unsigned short* base, int nbase,
                                        int kbase, int kstride) {
  ABPack p;
  const unsigned int* b32 = (const unsigned int*)base;
  const int lane = threadIdx.x & 31;
  const int n = nbase + (lane & 15);
  const int hi16 = (lane & 16) ? 16 : 0;
#pragma unroll
  for (int v = 0; v < 8; ++v) {
    const int k = kbase + 2 * v + hi16;
    p.u[v] = b32[(n * kstride + k) >> 1];
  }
  return p.v;
}

// Same B layout but storage is k-major [k][n] (used when only the transposed
// copy of K exists in LDS): two u16 loads + pack per dword.
__device__ __forceinline__ v16bf load_b_kmajor(const unsigned short* base, int nbase,
                                               int kbase, int nstride) {
  ABPack p;
  const int lane = threadIdx.x & 31;
  const int n = nbase + (lane & 15);
  const int hi16 = (lane & 16) ? 16 : 0;
#pragma unroll
  for (int v = 0; v < 8; ++v) {
    const int k = kbase + 2 * v + hi16;
    unsigned int lo = base[(k + 0) * nstride + n];
    unsigned int hh = base[(k + 1) * nstride + n];
    p.u[v] = lo | (hh << 16);
  }
  return p.v;
}

__device__ __forceinline__ v8f wmma_bf16(v16bf a, v16bf b, v8f c) {
  return __builtin_amdgcn_wmma_f32_16x16x32_bf16(false, a, false, b,
                                                 (short)0, c, false, false);
}

struct SMem {
  unsigned short Qrm[CHUNK * DMODEL];   // phi(Q) chunk, [t][f]        32 KB
  unsigned short KT [DMODEL * CHUNK];   // phi(K) chunk, [f][t]        32 KB
  unsigned short VT [DTILE * CHUNK];    // V chunk slice, [d][t]        4 KB
  unsigned short Arm[CHUNK * CHUNK];    // masked scores, [t][s]        2 KB
  unsigned short ST [DTILE * DMODEL];   // bf16 copy of state, [d][f]  64 KB
  float Z[DMODEL];                      // running key-sum (f32)
  float den[CHUNK];                     // per-row denominators
};                                      // ~136 KB < 320 KB/WGP

__global__ __launch_bounds__(NWAVE * 32, 1)
void linatt_chunked(const float* __restrict__ Q, const float* __restrict__ K,
                    const float* __restrict__ V, float* __restrict__ Out) {
  __shared__ SMem sm;

  const int tid  = threadIdx.x;
  const int wave = tid >> 5;
  const int lane = tid & 31;
  const int l16  = lane & 15;
  const int hi8  = (lane & 16) ? 8 : 0;

  const int b  = blockIdx.x >> 3;            // D/DTILE == 8 slices
  const int d0 = (blockIdx.x & 7) * DTILE;

  const float* Qb = Q   + (size_t)b * SEQ * DMODEL;
  const float* Kb = K   + (size_t)b * SEQ * DMODEL;
  const float* Vb = V   + (size_t)b * SEQ * DMODEL;
  float*       Ob = Out + (size_t)b * SEQ * DMODEL;

  // f32 state S[f][d_wave], 32 row tiles of 16x16 (C/D layout accumulators).
  v8f Sacc[32];
#pragma unroll
  for (int i = 0; i < 32; ++i) Sacc[i] = (v8f){};

  {   // zero bf16 state mirror and Z
    unsigned int* st32 = (unsigned int*)sm.ST;
    for (int i = tid; i < (DTILE * DMODEL / 2); i += NWAVE * 32) st32[i] = 0u;
    for (int i = tid; i < DMODEL; i += NWAVE * 32) sm.Z[i] = 0.0f;
  }
  __syncthreads();

  float denb = 0.0f;   // denominator base (valid on wave 1, lane == t)

  for (int c0 = 0; c0 < SEQ; c0 += CHUNK) {
    // ---------------- Phase 1: stage chunk into LDS (apply phi, cast bf16) --
    {
      const int fc = tid * 4;                         // one float4 column / thread
      unsigned int* q32 = (unsigned int*)sm.Qrm;
#pragma unroll 4
      for (int r = 0; r < CHUNK; ++r) {
        const float4 q4 = *(const float4*)(Qb + (size_t)(c0 + r) * DMODEL + fc);
        const float4 k4 = *(const float4*)(Kb + (size_t)(c0 + r) * DMODEL + fc);
        const unsigned short q0 = f2bf(phi_elu1(q4.x));
        const unsigned short q1 = f2bf(phi_elu1(q4.y));
        const unsigned short q2 = f2bf(phi_elu1(q4.z));
        const unsigned short q3 = f2bf(phi_elu1(q4.w));
        const int qi = (r * DMODEL + fc) >> 1;
        q32[qi + 0] = (unsigned int)q0 | ((unsigned int)q1 << 16);
        q32[qi + 1] = (unsigned int)q2 | ((unsigned int)q3 << 16);
        sm.KT[(fc + 0) * CHUNK + r] = f2bf(phi_elu1(k4.x));
        sm.KT[(fc + 1) * CHUNK + r] = f2bf(phi_elu1(k4.y));
        sm.KT[(fc + 2) * CHUNK + r] = f2bf(phi_elu1(k4.z));
        sm.KT[(fc + 3) * CHUNK + r] = f2bf(phi_elu1(k4.w));
        if (c0 + CHUNK < SEQ) {   // pull next chunk toward L2/L0
          __builtin_prefetch(Qb + (size_t)(c0 + CHUNK + r) * DMODEL + fc, 0, 1);
          __builtin_prefetch(Kb + (size_t)(c0 + CHUNK + r) * DMODEL + fc, 0, 1);
        }
      }
#pragma unroll
      for (int i = 0; i < 4; ++i) {           // V slice: 32 x 64, store transposed
        const int idx = i * 128 + tid;
        const int r   = idx >> 4;
        const int c4  = (idx & 15) * 4;
        const float4 v4 = *(const float4*)(Vb + (size_t)(c0 + r) * DMODEL + d0 + c4);
        sm.VT[(c4 + 0) * CHUNK + r] = f2bf(v4.x);
        sm.VT[(c4 + 1) * CHUNK + r] = f2bf(v4.y);
        sm.VT[(c4 + 2) * CHUNK + r] = f2bf(v4.z);
        sm.VT[(c4 + 3) * CHUNK + r] = f2bf(v4.w);
      }
    }
    __syncthreads();

    // ---------------- Phase 2: A = tril(Q K^T); denominator base -----------
    {
      const int ti = (wave >> 1) * 16;        // score tile row block
      const int sj = (wave & 1)  * 16;        // score tile col block
      if (!(ti == 0 && sj == 16)) {           // (0,1) is entirely above diagonal
        v8f a_acc = (v8f){};
#pragma unroll
        for (int kb = 0; kb < DMODEL; kb += 32) {
          v16bf av = load_a(sm.Qrm, ti, kb, DMODEL);
          v16bf bv = load_b_kmajor(sm.KT, sj, kb, CHUNK);
          a_acc = wmma_bf16(av, bv, a_acc);
        }
#pragma unroll
        for (int r = 0; r < 8; ++r) {         // causal mask + bf16 store [t][s]
          const int t = ti + r + hi8;
          const int s = sj + l16;
          const float m = (s <= t) ? a_acc[r] : 0.0f;
          sm.Arm[t * CHUNK + s] = f2bf(m);
        }
      } else {
        // fully-masked tile's wave: zero its region, compute q_t . Z_prev
#pragma unroll
        for (int r = 0; r < 8; ++r) sm.Arm[(ti + r + hi8) * CHUNK + sj + l16] = 0;
        float acc = 0.0f;
        const unsigned int* qrow = (const unsigned int*)(sm.Qrm + lane * DMODEL);
        for (int fw = 0; fw < DMODEL / 2; ++fw) {
          const unsigned int u = qrow[fw];
          acc += bf2f((unsigned short)(u & 0xFFFFu)) * sm.Z[2 * fw + 0];
          acc += bf2f((unsigned short)(u >> 16))     * sm.Z[2 * fw + 1];
        }
        denb = acc;
      }
    }
    __syncthreads();

    // ---------------- Phase 3: Z += sum_k; den[t] = base + rowsum(A) -------
    {
#pragma unroll
      for (int j = 0; j < 4; ++j) {
        const int f = tid * 4 + j;
        const unsigned int* kr = (const unsigned int*)(sm.KT + f * CHUNK);
        float s = 0.0f;
#pragma unroll
        for (int w = 0; w < CHUNK / 2; ++w) {
          const unsigned int u = kr[w];
          s += bf2f((unsigned short)(u & 0xFFFFu)) + bf2f((unsigned short)(u >> 16));
        }
        sm.Z[f] += s;
      }
      if (wave == 1) {                        // lane == t
        float rs = denb;
        for (int s = 0; s <= lane; ++s) rs += bf2f(sm.Arm[lane * CHUNK + s]);
        sm.den[lane] = fmaxf(rs, EPS_DEN);
      }
    }
    __syncthreads();

    // ---------------- Phase 4: O = Q*S_prev + A*V; S += K^T V --------------
    const int dw = wave * 16;                 // this wave's 16 D-columns
#pragma unroll
    for (int ti = 0; ti < 2; ++ti) {
      v8f o = (v8f){};
#pragma unroll
      for (int kb = 0; kb < DMODEL; kb += 32) {       // inter-chunk: Q @ S_prev
        v16bf av = load_a(sm.Qrm, ti * 16, kb, DMODEL);
        v16bf bv = load_b(sm.ST + (size_t)dw * DMODEL, 0, kb, DMODEL);
        o = wmma_bf16(av, bv, o);
      }
      {                                               // intra-chunk: A @ V
        v16bf av = load_a(sm.Arm, ti * 16, 0, CHUNK);
        v16bf bv = load_b(sm.VT, dw, 0, CHUNK);
        o = wmma_bf16(av, bv, o);
      }
#pragma unroll
      for (int r = 0; r < 8; ++r) {                   // normalize + store
        const int t = ti * 16 + r + hi8;
        Ob[(size_t)(c0 + t) * DMODEL + d0 + dw + l16] = o[r] / sm.den[t];
      }
    }

    // state update in f32 accumulators: S += K^T @ V
#pragma unroll
    for (int fr = 0; fr < 32; ++fr) {
      v16bf av = load_a(sm.KT, fr * 16, 0, CHUNK);
      v16bf bv = load_b(sm.VT, dw, 0, CHUNK);
      Sacc[fr] = wmma_bf16(av, bv, Sacc[fr]);
    }

    // publish bf16 mirror of S (own region only -> no extra barrier needed)
#pragma unroll
    for (int fr = 0; fr < 32; ++fr) {
#pragma unroll
      for (int r = 0; r < 8; ++r) {
        const int f = fr * 16 + r + hi8;
        sm.ST[(dw + l16) * DMODEL + f] = f2bf(Sacc[fr][r]);
      }
    }
    __syncthreads();   // protect Qrm/KT/VT/Arm/den/Z before next stage
  }
}

extern "C" void kernel_launch(void* const* d_in, const int* in_sizes, int n_in,
                              void* d_out, int out_size, void* d_ws, size_t ws_size,
                              hipStream_t stream) {
  (void)in_sizes; (void)n_in; (void)out_size; (void)d_ws; (void)ws_size;
  const float* Q = (const float*)d_in[0];
  const float* K = (const float*)d_in[1];
  const float* V = (const float*)d_in[2];
  float* Out = (float*)d_out;
  dim3 grid(BATCH * (DMODEL / DTILE));   // 32 workgroups
  dim3 block(NWAVE * 32);                // 4 wave32
  linatt_chunked<<<grid, block, 0, stream>>>(Q, K, V, Out);
}
